// GRU_30365418782870
// MI455X (gfx1250) — matrix operand order repaired and checked by reading, not compiled, not run on hardware
//
#include <hip/hip_runtime.h>
#include <hip/hip_bf16.h>

#define EMB 512
#define HID 512
#define SEQ 128
#define BATCH 64
#define VOCAB 10000
#define VPAD 10240          // VOCAB padded to a multiple of BN (store-guarded)

typedef __bf16 bf16;
typedef __attribute__((ext_vector_type(16))) __bf16 v16bf;
typedef __attribute__((ext_vector_type(8)))  __bf16 v8bf;
typedef __attribute__((ext_vector_type(8)))  float  v8f;

union Frag { v16bf v; v8bf h[2]; };

__device__ __forceinline__ bf16 f2bf(float f) {
  unsigned u = __float_as_uint(f);
  u += 0x7FFFu + ((u >> 16) & 1u);              // round-to-nearest-even
  unsigned short s = (unsigned short)(u >> 16);
  return __builtin_bit_cast(__bf16, s);
}

// ---- CDNA5 async global->LDS copy (ASYNCcnt-tracked) -----------------------
__device__ __forceinline__ void async_b128(unsigned lds_off, const void* gptr) {
  unsigned long long ga = (unsigned long long)gptr;
  asm volatile("global_load_async_to_lds_b128 %0, %1, off"
               :: "v"(lds_off), "v"(ga) : "memory");
}
__device__ __forceinline__ void wait_async6() {
  asm volatile("s_wait_asynccnt 6" ::: "memory");
}
__device__ __forceinline__ void wait_async0() {
  asm volatile("s_wait_asynccnt 0" ::: "memory");
}
__device__ __forceinline__ unsigned lds_off(const void* p) {
  return (unsigned)(unsigned long long)p;       // generic LDS addr: low 32 bits = offset
}

// ===========================================================================
// Big LDS-staged GEMM: C[M,N] = A[M,K] @ WT[N,K]^T + bias, bf16 in / f32 out.
// Block (256 thr, 8 waves) computes BM x BN; k-chunks of 32 double-buffered in
// LDS via async copies. Each wave: 64x64 sub-tile = 4x4 wmma per chunk.
// ===========================================================================
#define BM 128
#define BN 256
#define PADROW 40           // LDS row pitch in bf16 (80 B: 16B-aligned, bank-spread)

__device__ __forceinline__ void issue_chunk(
    const bf16* __restrict__ A, int lda,
    const bf16* __restrict__ WT, int ldw,
    int m0, int n0, int kc, int tid,
    unsigned ldsA, unsigned ldsB)
{
  // A panel: BM x 32 bf16 = 512 x 16B segments -> 2 per thread
#pragma unroll
  for (int j = 0; j < 2; ++j) {
    int seg = tid + 256 * j;
    int row = seg >> 2, part = seg & 3;
    async_b128(ldsA + (unsigned)(row * PADROW + part * 8) * 2,
               A + (size_t)(m0 + row) * lda + kc + part * 8);
  }
  // B panel: BN x 32 bf16 = 1024 x 16B segments -> 4 per thread
#pragma unroll
  for (int j = 0; j < 4; ++j) {
    int seg = tid + 256 * j;
    int row = seg >> 2, part = seg & 3;
    async_b128(ldsB + (unsigned)(row * PADROW + part * 8) * 2,
               WT + (size_t)(n0 + row) * ldw + kc + part * 8);
  }
}

__global__ __launch_bounds__(256)
void gemm_lds_kernel(const bf16* __restrict__ A, int lda,
                     const bf16* __restrict__ WT, int ldw,
                     const float* __restrict__ bias,
                     float* __restrict__ C, int ldc,
                     int M, int N, int kLen)
{
  __shared__ __align__(16) bf16 sA[2][BM * PADROW];   // 2 x 10 KB
  __shared__ __align__(16) bf16 sB[2][BN * PADROW];   // 2 x 20 KB
  int nBlocks = (N + BN - 1) / BN;
  int m0 = (blockIdx.x / nBlocks) * BM;
  int n0 = (blockIdx.x % nBlocks) * BN;
  int tid = threadIdx.x;
  int lane = tid & 31, l15 = lane & 15, hi = lane >> 4;
  int wv = tid >> 5;
  int wm = (wv & 1) * 64;       // wave sub-tile origin inside block tile
  int wn = (wv >> 1) * 64;

  unsigned lA[2] = { lds_off(&sA[0][0]), lds_off(&sA[1][0]) };
  unsigned lB[2] = { lds_off(&sB[0][0]), lds_off(&sB[1][0]) };

  v8f acc[4][4] = {};

  issue_chunk(A, lda, WT, ldw, m0, n0, 0, tid, lA[0], lB[0]);

  const int nChunks = kLen >> 5;
  for (int i = 0; i < nChunks; ++i) {
    int cur = i & 1;
    bool more = (i + 1) < nChunks;
    if (more)
      issue_chunk(A, lda, WT, ldw, m0, n0, (i + 1) * 32, tid, lA[cur ^ 1], lB[cur ^ 1]);
    if (more) wait_async6(); else wait_async0();   // chunk i's copies landed
    __syncthreads();                                // ...for every wave

    const bf16* Ab = &sA[cur][0];
    const bf16* Bb = &sB[cur][0];
    Frag af[4], bfr[4];
#pragma unroll
    for (int t = 0; t < 4; ++t) {                   // A frags (ds_load_b128 x2)
      const bf16* r = Ab + (wm + t * 16 + l15) * PADROW + 8 * hi;
      af[t].h[0] = *(const v8bf*)r;
      af[t].h[1] = *(const v8bf*)(r + 16);
    }
#pragma unroll
    for (int t = 0; t < 4; ++t) {                   // B frags
      const bf16* r = Bb + (wn + t * 16 + l15) * PADROW + 8 * hi;
      bfr[t].h[0] = *(const v8bf*)r;
      bfr[t].h[1] = *(const v8bf*)(r + 16);
    }
#pragma unroll
    for (int tm = 0; tm < 4; ++tm)
#pragma unroll
      for (int tn = 0; tn < 4; ++tn)
        acc[tm][tn] = __builtin_amdgcn_wmma_f32_16x16x32_bf16(
            false, af[tm].v, false, bfr[tn].v, (short)0, acc[tm][tn], false, false);
    __syncthreads();                                // buffer reusable
  }

#pragma unroll
  for (int tm = 0; tm < 4; ++tm)
#pragma unroll
    for (int tn = 0; tn < 4; ++tn) {
      int n = n0 + wn + tn * 16 + l15;
      if (n < N) {
        float bn = bias ? bias[n] : 0.f;
#pragma unroll
        for (int r = 0; r < 8; ++r) {
          int m = m0 + wm + tm * 16 + r + 8 * hi;
          C[(size_t)m * ldc + n] = acc[tm][tn][r] + bn;
        }
      }
    }
}

// ===========================================================================
// Software-pipelined wave GEMM core for the small recurrent GEMMs (M=64).
// Next chunk's fragments are issued before the current chunk's wmma so the
// loads retire behind the matrix op instead of stalling in front of it.
// ===========================================================================
template<int MT>
__device__ __forceinline__ void gemm_core(
    const bf16* __restrict__ A0, const bf16* __restrict__ A1, int split, int lda,
    const bf16* __restrict__ WT, int ldw, int bk0,
    int m0, int n0, int kLen, v8f acc[MT])
{
  const int lane = threadIdx.x & 31;
  const int l15  = lane & 15;
  const int hi   = lane >> 4;
  Frag b, a[MT];
  {
    const bf16* br = WT + (size_t)(n0 + l15) * ldw + (bk0 + 8 * hi);
    b.h[0] = *(const v8bf*)br; b.h[1] = *(const v8bf*)(br + 16);
#pragma unroll
    for (int t = 0; t < MT; ++t) {
      const bf16* ar = A0 + (size_t)(m0 + t * 16 + l15) * lda + 8 * hi;
      a[t].h[0] = *(const v8bf*)ar; a[t].h[1] = *(const v8bf*)(ar + 16);
    }
  }
  for (int kc = 0; kc < kLen; kc += 32) {
    Frag b2, a2[MT];
    int kn = kc + 32;
    if (kn < kLen) {
      const bf16* br = WT + (size_t)(n0 + l15) * ldw + (bk0 + kn + 8 * hi);
      b2.h[0] = *(const v8bf*)br; b2.h[1] = *(const v8bf*)(br + 16);
      const bf16* Ap = (kn < split) ? A0 : A1;
      int kA = (kn < split) ? kn : kn - split;
#pragma unroll
      for (int t = 0; t < MT; ++t) {
        const bf16* ar = Ap + (size_t)(m0 + t * 16 + l15) * lda + kA + 8 * hi;
        a2[t].h[0] = *(const v8bf*)ar; a2[t].h[1] = *(const v8bf*)(ar + 16);
      }
    }
#pragma unroll
    for (int t = 0; t < MT; ++t)
      acc[t] = __builtin_amdgcn_wmma_f32_16x16x32_bf16(
          false, a[t].v, false, b.v, (short)0, acc[t], false, false);
    b = b2;
#pragma unroll
    for (int t = 0; t < MT; ++t) a[t] = a2[t];
  }
}

// ---- Fused gate kernel: g = sigmoid(acc + pre|bias); writes update gate U
// ---- and bf16 reset*h (RH) for the following cell GEMM --------------------
__global__ __launch_bounds__(256)
void gru_gates_kernel(const bf16* __restrict__ A0, const bf16* __restrict__ A1,
                      int split, int lda,
                      const bf16* __restrict__ WT, int ldw, int bk0, int kLen,
                      const float* __restrict__ pre, const float* __restrict__ bias,
                      const float* __restrict__ hstate,
                      float* __restrict__ U, bf16* __restrict__ RH)
{
  int wave = blockIdx.x * (blockDim.x >> 5) + (threadIdx.x >> 5);
  const int mTiles = BATCH / 16;          // 4
  const int nTiles = (2 * HID) / 16;      // 64
  if (wave >= mTiles * nTiles) return;
  int m0 = (wave % mTiles) << 4;
  int n0 = (wave / mTiles) << 4;
  v8f acc[1] = {};
  gemm_core<1>(A0, A1, split, lda, WT, ldw, bk0, m0, n0, kLen, acc);
  int lane = threadIdx.x & 31;
  int n  = n0 + (lane & 15);
  int hi = lane >> 4;
  float bn = bias ? bias[n] : 0.f;
#pragma unroll
  for (int r = 0; r < 8; ++r) {
    int m = m0 + r + 8 * hi;
    float x = acc[0][r] + bn + (pre ? pre[(size_t)m * (2 * HID) + n] : 0.f);
    float g = 1.f / (1.f + __expf(-x));
    if (n < HID) {
      U[(size_t)m * HID + n] = g;                       // update gate
    } else {
      int nn = n - HID;                                 // reset gate * h
      RH[(size_t)m * HID + nn] = f2bf(g * hstate[(size_t)m * HID + nn]);
    }
  }
}

// ---- Fused cell kernel: c = tanh(acc + pre|bias); h' = (1-u)h + u*c --------
__global__ __launch_bounds__(256)
void gru_cell_kernel(const bf16* __restrict__ A0, const bf16* __restrict__ A1,
                     int split, int lda,
                     const bf16* __restrict__ WT, int ldw, int bk0, int kLen,
                     const float* __restrict__ pre, const float* __restrict__ bias,
                     const float* __restrict__ U,
                     float* __restrict__ hf, bf16* __restrict__ hb,
                     bf16* __restrict__ Hout)
{
  int wave = blockIdx.x * (blockDim.x >> 5) + (threadIdx.x >> 5);
  const int mTiles = BATCH / 16;          // 4
  const int nTiles = HID / 16;            // 32
  if (wave >= mTiles * nTiles) return;
  int m0 = (wave % mTiles) << 4;
  int n0 = (wave / mTiles) << 4;
  v8f acc[1] = {};
  gemm_core<1>(A0, A1, split, lda, WT, ldw, bk0, m0, n0, kLen, acc);
  int lane = threadIdx.x & 31;
  int n  = n0 + (lane & 15);
  int hi = lane >> 4;
  float bn = bias ? bias[n] : 0.f;
#pragma unroll
  for (int r = 0; r < 8; ++r) {
    int m = m0 + r + 8 * hi;
    size_t idx = (size_t)m * HID + n;
    float x = acc[0][r] + bn + (pre ? pre[idx] : 0.f);
    float c = tanhf(x);
    float u = U[idx];
    float hnew = (1.f - u) * hf[idx] + u * c;
    hf[idx] = hnew;
    bf16 hv = f2bf(hnew);
    hb[idx] = hv;
    if (Hout) Hout[idx] = hv;
  }
}

// ---- One-time prep kernels -------------------------------------------------
__global__ void wt_kernel(const float* __restrict__ W, bf16* __restrict__ WT,
                          int K, int N) {           // W is (K,N) -> WT is (N,K) bf16
  size_t gid = (size_t)blockIdx.x * blockDim.x + threadIdx.x;
  if (gid >= (size_t)K * N) return;
  int n = (int)(gid / K), k = (int)(gid % K);
  WT[gid] = f2bf(W[(size_t)k * N + n]);
}

__global__ void embed_kernel(const int* __restrict__ inputs,
                             const float* __restrict__ emb,
                             bf16* __restrict__ X) {
  int row = blockIdx.x;                              // s*BATCH + b
  int tok = inputs[row];
  const float s = 22.62741699796952f;                // sqrt(EMB)
  for (int e = threadIdx.x; e < EMB; e += blockDim.x)
    X[(size_t)row * EMB + e] = f2bf(emb[(size_t)tok * EMB + e] * s);
}

__global__ void init_state_kernel(const float* __restrict__ hidden,
                                  float* h0f, float* h1f, bf16* h0b, bf16* h1b) {
  int i = blockIdx.x * blockDim.x + threadIdx.x;
  if (i >= BATCH * HID) return;
  float a = hidden[i], b = hidden[BATCH * HID + i];
  h0f[i] = a; h1f[i] = b; h0b[i] = f2bf(a); h1b[i] = f2bf(b);
}

__global__ void final_hidden_kernel(const float* h0f, const float* h1f,
                                    float* __restrict__ out) {
  int i = blockIdx.x * blockDim.x + threadIdx.x;
  if (i >= BATCH * HID) return;
  out[i] = h0f[i];
  out[BATCH * HID + i] = h1f[i];
}

// ---------------------------------------------------------------------------
extern "C" void kernel_launch(void* const* d_in, const int* in_sizes, int n_in,
                              void* d_out, int out_size, void* d_ws, size_t ws_size,
                              hipStream_t stream) {
  const int*   inputs    = (const int*)  d_in[0];
  const float* hidden    = (const float*)d_in[1];
  const float* emb_table = (const float*)d_in[2];
  const float* Wg0 = (const float*)d_in[3];  const float* bg0 = (const float*)d_in[4];
  const float* Wc0 = (const float*)d_in[5];  const float* bc0 = (const float*)d_in[6];
  const float* Wg1 = (const float*)d_in[7];  const float* bg1 = (const float*)d_in[8];
  const float* Wc1 = (const float*)d_in[9];  const float* bc1 = (const float*)d_in[10];
  const float* Wout = (const float*)d_in[11]; const float* bout = (const float*)d_in[12];
  float* out = (float*)d_out;

  char* p = (char*)d_ws;
  auto take = [&](size_t nbytes) -> char* {
    char* r = p; p += (nbytes + 255) & ~(size_t)255; return r;
  };
  bf16*  WT_g0 = (bf16*) take((size_t)(2*HID) * (EMB+HID) * 2);
  bf16*  WT_c0 = (bf16*) take((size_t)HID     * (EMB+HID) * 2);
  bf16*  WT_g1 = (bf16*) take((size_t)(2*HID) * (2*HID)   * 2);
  bf16*  WT_c1 = (bf16*) take((size_t)HID     * (2*HID)   * 2);
  bf16*  WT_o  = (bf16*) take((size_t)VPAD    * HID       * 2);   // padded rows
  bf16*  X     = (bf16*) take((size_t)SEQ*BATCH*EMB * 2);
  bf16*  H1    = (bf16*) take((size_t)SEQ*BATCH*HID * 2);
  float* Xg0   = (float*)take((size_t)SEQ*BATCH*(2*HID) * 4);
  float* Xc0   = (float*)take((size_t)SEQ*BATCH*HID * 4);
  float* h0f   = (float*)take((size_t)BATCH*HID * 4);
  float* h1f   = (float*)take((size_t)BATCH*HID * 4);
  bf16*  h0b   = (bf16*) take((size_t)BATCH*HID * 2);
  bf16*  h1b   = (bf16*) take((size_t)BATCH*HID * 2);
  float* G0u   = (float*)take((size_t)BATCH*HID * 4);
  float* G1u   = (float*)take((size_t)BATCH*HID * 4);
  bf16*  RH0   = (bf16*) take((size_t)BATCH*HID * 2);
  bf16*  RH1   = (bf16*) take((size_t)BATCH*HID * 2);

  const int BIG = 1 << 30;

  // --- weight transpose + bf16 convert (weights then stay L2-resident)
  wt_kernel<<<(1024*1024 + 255) / 256, 256, 0, stream>>>(Wg0, WT_g0, EMB+HID, 2*HID);
  wt_kernel<<<(1024*512  + 255) / 256, 256, 0, stream>>>(Wc0, WT_c0, EMB+HID, HID);
  wt_kernel<<<(1024*1024 + 255) / 256, 256, 0, stream>>>(Wg1, WT_g1, 2*HID,   2*HID);
  wt_kernel<<<(1024*512  + 255) / 256, 256, 0, stream>>>(Wc1, WT_c1, 2*HID,   HID);
  wt_kernel<<<(512*VOCAB + 255) / 256, 256, 0, stream>>>(Wout, WT_o, HID,     VOCAB);

  // --- embeddings + state init
  embed_kernel<<<SEQ*BATCH, 256, 0, stream>>>(inputs, emb_table, X);
  init_state_kernel<<<(BATCH*HID + 255) / 256, 256, 0, stream>>>(hidden, h0f, h1f, h0b, h1b);

  // --- hoisted layer-0 input projections (bias folded), LDS-staged WMMA GEMMs
  gemm_lds_kernel<<<(SEQ*BATCH/BM) * (2*HID/BN), 256, 0, stream>>>(
      X, EMB, WT_g0, EMB+HID, bg0, Xg0, 2*HID, SEQ*BATCH, 2*HID, EMB);
  gemm_lds_kernel<<<(SEQ*BATCH/BM) * (HID/BN), 256, 0, stream>>>(
      X, EMB, WT_c0, EMB+HID, bc0, Xc0, HID, SEQ*BATCH, HID, EMB);

  // --- sequential recurrence: 4 fused GEMM kernels per step
  for (int t = 0; t < SEQ; ++t) {
    const float* pg = Xg0 + (size_t)t * BATCH * (2*HID);
    const float* pc = Xc0 + (size_t)t * BATCH * HID;
    gru_gates_kernel<<<32, 256, 0, stream>>>(h0b, h0b, BIG, HID,
                                             WT_g0, EMB+HID, EMB, HID,
                                             pg, nullptr, h0f, G0u, RH0);
    gru_cell_kernel<<<16, 256, 0, stream>>>(RH0, RH0, BIG, HID,
                                            WT_c0, EMB+HID, EMB, HID,
                                            pc, nullptr, G0u, h0f, h0b, nullptr);
    gru_gates_kernel<<<32, 256, 0, stream>>>(h0b, h1b, HID, HID,
                                             WT_g1, 2*HID, 0, 2*HID,
                                             nullptr, bg1, h1f, G1u, RH1);
    gru_cell_kernel<<<16, 256, 0, stream>>>(h0b, RH1, HID, HID,
                                            WT_c1, 2*HID, 0, 2*HID,
                                            nullptr, bc1, G1u, h1f, h1b,
                                            H1 + (size_t)t * BATCH * HID);
  }

  // --- batched vocab projection: logits = H1 @ Wout + bout (8192x10000x512)
  gemm_lds_kernel<<<(SEQ*BATCH/BM) * (VPAD/BN), 256, 0, stream>>>(
      H1, HID, WT_o, HID, bout, out, VOCAB, SEQ*BATCH, VOCAB, HID);

  // --- final hidden (2, BATCH, HID) appended after logits
  final_hidden_kernel<<<(BATCH*HID + 255) / 256, 256, 0, stream>>>(
      h0f, h1f, out + (size_t)SEQ * BATCH * VOCAB);
}
